// MultiHeadCrossAttention_70377334112919
// MI455X (gfx1250) — compile-verified, matrix operand order checked by reading, not run on hardware
//
#include <hip/hip_runtime.h>

// MI455X / gfx1250, wave32. All GEMM stages use v_wmma_f32_16x16x32_f16
// (f16 A/B, f32 accumulate). Working set (~53 MB) fits in the 192 MB L2.
// Shared B operands (packed weights / vh slabs / kh row strips) are staged
// into LDS once per block with the Tensor Data Mover (TENSOR_LOAD_TO_LDS +
// s_wait_tensorcnt), then read as ds_load_b128 fragments; A operands stream
// from L2 with global_prefetch_b8.

typedef __attribute__((ext_vector_type(16))) _Float16 v16h;
typedef __attribute__((ext_vector_type(8)))  _Float16 v8h;
typedef __attribute__((ext_vector_type(8)))  float    v8f;
typedef __attribute__((ext_vector_type(4)))  unsigned int u32x4;
typedef __attribute__((ext_vector_type(8)))  int      i32x8;
typedef __attribute__((ext_vector_type(4)))  int      i32x4;

#define BSZ 4
#define SEQ 512
#define DMODEL 512
#define NH 8
#define DKH 64
#define MROWS (BSZ * SEQ)   // 2048
#define EPS 1e-6f

// ---------------------------------------------------------------------------
// Tensor Data Mover: 1-D contiguous copy global -> LDS (ISA ch. 8 D# layout).
// nelem = f16 element count (<= 65535 to fit 16-bit tile_dim0).
// Caller: issue from ONE wave (TDM ignores EXEC; one issue per wave), then
// s_wait_tensorcnt(0) on all waves + __syncthreads().
// ---------------------------------------------------------------------------
static __device__ inline void tdm_load_1d_to_lds(const _Float16* gsrc,
                                                 unsigned lds_addr,
                                                 unsigned nelem) {
  const unsigned long long ga = (unsigned long long)gsrc;
  u32x4 g0 = {};
  g0[0] = 1u;                                            // count=1 user D#
  g0[1] = lds_addr;                                      // lds_addr (bytes)
  g0[2] = (unsigned)ga;                                  // global_addr[31:0]
  g0[3] = (unsigned)((ga >> 32) & 0x01FFFFFFu) | (2u << 30); // addr[56:32]|type=2
  i32x8 g1 = {};
  g1[0] = (int)(1u << 16);                               // data_size = 2 bytes
  g1[1] = (int)((nelem & 0xFFFFu) << 16);                // tensor_dim0[15:0]
  g1[2] = (int)((nelem >> 16) | (1u << 16));             // tensor_dim0 hi, tensor_dim1=1
  g1[3] = (int)((nelem & 0xFFFFu) << 16);                // tile_dim0 = nelem
  g1[4] = 0;                                             // tile_dim1/2 unused
  g1[5] = (int)nelem;                                    // tensor_dim0_stride
  g1[6] = 0;
  g1[7] = 0;
  i32x4 gz = {};
#if __clang_major__ >= 23
  i32x8 gz8 = {};
  __builtin_amdgcn_tensor_load_to_lds(g0, g1, gz, gz, gz8, 0);
#else
  __builtin_amdgcn_tensor_load_to_lds(g0, g1, gz, gz, 0);
#endif
}

static __device__ inline unsigned lds_addr_of(const void* p) {
  return (unsigned)(unsigned long long)p;   // LDS offset for D#.lds_addr
}

// ---------------------------------------------------------------------------
// WMMA fragment helpers (ISA 7.12.2 layouts, wave32)
// ---------------------------------------------------------------------------

static __device__ inline v16h cat8(v8h lo, v8h hi) {
  return __builtin_shufflevector(lo, hi, 0, 1, 2, 3, 4, 5, 6, 7,
                                 8, 9, 10, 11, 12, 13, 14, 15);
}

// A-matrix 16x32 f16 fragment from row-major A[row][k] (lda element stride).
// Per lane: two contiguous 16B chunks. Also loads B^T fragments (Q @ K^T).
static __device__ inline v16h load_af(const _Float16* __restrict__ A, int lda) {
  const int lane = threadIdx.x & 31;
  const int row  = lane & 15;
  const int khi  = (lane >> 4) << 3;
  const _Float16* p = A + (size_t)row * lda + khi;
  return cat8(*(const v8h*)p, *(const v8h*)(p + 16));
}

// B-matrix 32x16 fragment from the packed layout (512 halves per tile,
// lane-major): one 32B contiguous load per lane. Works for global or LDS
// pointers (LDS -> ds_load_b128 x2).
static __device__ inline v16h load_bp(const _Float16* P, int tile) {
  const int lane = threadIdx.x & 31;
  return *(const v16h*)(P + ((size_t)tile << 9) + (lane << 4));
}

// C/D 16x16 f32: VGPR r, lane L -> M = r + 8*(L>>4), N = L&15.
static __device__ inline void store_d_f32(float* __restrict__ C, int ldc, v8f d) {
  const int lane = threadIdx.x & 31;
  const int n    = lane & 15;
  const int mhi  = (lane >> 4) << 3;
#pragma unroll
  for (int r = 0; r < 8; ++r)
    C[(size_t)(mhi + r) * ldc + n] = d[r];
}

static __device__ inline void store_d_f16(_Float16* __restrict__ C, int ldc, v8f d) {
  const int lane = threadIdx.x & 31;
  const int n    = lane & 15;
  const int mhi  = (lane >> 4) << 3;
#pragma unroll
  for (int r = 0; r < 8; ++r)
    C[(size_t)(mhi + r) * ldc + n] = (_Float16)d[r];
}

static __device__ inline v8f wmma_f16(v16h a, v16h b, v8f c) {
  return __builtin_amdgcn_wmma_f32_16x16x32_f16(
      /*neg_a=*/false, a, /*neg_b=*/false, b,
      /*c_mod=*/(short)0, c, /*reuse_a=*/false, /*reuse_b=*/false);
}

// ---------------------------------------------------------------------------
// Kernel 0: f32 -> f16 convert (q, k, v activations)
// ---------------------------------------------------------------------------
__global__ void k_f32_to_f16(const float* __restrict__ src,
                             _Float16* __restrict__ dst, int n) {
  int i = blockIdx.x * blockDim.x + threadIdx.x;
  if (i < n) dst[i] = (_Float16)src[i];
}

// ---------------------------------------------------------------------------
// Pack kernels: scatter-gather a row-major [K][N] matrix into the WMMA
// B-fragment-linear layout. Tile id = nt*(K/32) + kt; within a tile,
// dst[tile*512 + lane*16 + e] = B[k(e)][n(lane)].
// ---------------------------------------------------------------------------
static __device__ inline void pack_coords(int idx, int KT, int* kk, int* nn) {
  const int tile = idx >> 9;
  const int rem  = idx & 511;
  const int lane = rem >> 4;
  const int e    = rem & 15;
  const int kt = tile % KT;
  const int nt = tile / KT;
  const int u = e >> 1;
  *nn = nt * 16 + (lane & 15);
  *kk = kt * 32 + ((u < 4) ? 0 : 16) + ((lane >> 4) << 3) + 2 * (u & 3) + (e & 1);
}

__global__ void k_pack_b_f32(const float* __restrict__ src,
                             _Float16* __restrict__ dst, int K, int N) {
  const int idx = blockIdx.x * blockDim.x + threadIdx.x;
  if (idx >= K * N) return;
  int k, n;
  pack_coords(idx, K >> 5, &k, &n);
  dst[idx] = (_Float16)src[(size_t)k * N + n];
}

// Batched over grid.y slabs (used for vh: 32 slabs of [S][DK]).
__global__ void k_pack_b_f16(const _Float16* __restrict__ src,
                             _Float16* __restrict__ dst, int K, int N,
                             size_t slab_elems) {
  const int idx = blockIdx.x * blockDim.x + threadIdx.x;
  if (idx >= K * N) return;
  const size_t off = (size_t)blockIdx.y * slab_elems;
  int k, n;
  pack_coords(idx, K >> 5, &k, &n);
  dst[off + idx] = src[off + (size_t)k * N + n];
}

// ---------------------------------------------------------------------------
// Kernel 1: projection GEMM  P = X(2048x512) @ W(512x512) + bias,
// W packed; the block's 64-col strip (64 tiles = 64KB) is TDM-staged to LDS.
// Block: 256 thr = 8 waves (4 M x 2 N), block tile 128x64, 2x2 frags/wave.
// Epilogue scatters to head layout [B,H,S,DK].
// ---------------------------------------------------------------------------
template <bool WRITE32>
__global__ __launch_bounds__(256) void k_gemm_proj_head(
    const _Float16* __restrict__ X, const _Float16* __restrict__ Wp,
    const float* __restrict__ bias,
    _Float16* __restrict__ out16, float* __restrict__ out32) {
  constexpr int KT = DMODEL >> 5;                 // 16 k-tiles
  __shared__ __align__(32) _Float16 ldsB[4 * KT * 512];   // 64KB

  const int w  = threadIdx.x >> 5;
  const int m0 = blockIdx.x * 128 + (w & 3) * 32;
  const int n0 = blockIdx.y * 64 + (w >> 2) * 32;
  const int ltile0 = ((w >> 2) * 2) * KT;         // wave's first local n-tile

  if (w == 0)                                     // one TDM issue per block
    tdm_load_1d_to_lds(Wp + ((size_t)(blockIdx.y * 4) * KT << 9),
                       lds_addr_of(ldsB), 4 * KT * 512);
  __builtin_amdgcn_s_wait_tensorcnt(0);
  __syncthreads();

  v8f acc[2][2] = {{{}, {}}, {{}, {}}};
  for (int kt = 0; kt < KT; ++kt) {
    const int k0 = kt << 5;
    if (kt + 1 < KT) {                            // gfx1250 global_prefetch_b8
      __builtin_prefetch(X + (size_t)m0 * DMODEL + k0 + 32, 0, 1);
      __builtin_prefetch(X + (size_t)(m0 + 16) * DMODEL + k0 + 32, 0, 1);
    }
    v16h a0 = load_af(X + (size_t)m0 * DMODEL + k0, DMODEL);
    v16h a1 = load_af(X + (size_t)(m0 + 16) * DMODEL + k0, DMODEL);
    v16h b0 = load_bp(ldsB, ltile0 + kt);
    v16h b1 = load_bp(ldsB, ltile0 + KT + kt);
    acc[0][0] = wmma_f16(a0, b0, acc[0][0]);
    acc[0][1] = wmma_f16(a0, b1, acc[0][1]);
    acc[1][0] = wmma_f16(a1, b0, acc[1][0]);
    acc[1][1] = wmma_f16(a1, b1, acc[1][1]);
  }

  const int lane = threadIdx.x & 31;
  const int nl   = lane & 15;
  const int mhi  = (lane >> 4) << 3;
#pragma unroll
  for (int i = 0; i < 2; ++i)
#pragma unroll
    for (int j = 0; j < 2; ++j) {
      const int col = n0 + 16 * j + nl;
      const float bv = bias[col];
      const int h = col >> 6, dd = col & 63;
#pragma unroll
      for (int r = 0; r < 8; ++r) {
        const int row = m0 + 16 * i + mhi + r;
        const int b = row >> 9, s = row & 511;
        const float val = acc[i][j][r] + bv;
        const size_t off = (((size_t)(b * NH + h) * SEQ) + s) * DKH + dd;
        out16[off] = (_Float16)val;
        if constexpr (WRITE32) out32[off] = val;
      }
    }
}

// ---------------------------------------------------------------------------
// Kernel 2: R tiles.  For each (b,h), 16x16 tile (i0,j0):
//   E_c = exp(qh[b,h] @ kh[c,h]^T), c = 0..3 (DK=64 -> 2 WMMA k-steps)
//   R   = E_b / (sqrt(64) * sum_c E_c), stored f16 row-major [S][S].
// The 4 kh row strips (64 rows x 64) used by the block are TDM-staged (32KB).
// Block 128 thr = 4 waves, one j-tile per wave. Grid (S/64, S/16, B*H).
// ---------------------------------------------------------------------------
__global__ __launch_bounds__(128) void k_pairwise_R(
    const _Float16* __restrict__ qh16, const _Float16* __restrict__ kh16,
    _Float16* __restrict__ R16) {
  __shared__ __align__(32) _Float16 ldsK[BSZ][64 * DKH];  // 4 x 8KB

  const int w   = threadIdx.x >> 5;
  const int jblk = blockIdx.x * 64;
  const int j0l  = w * 16;                 // j offset within the block strip
  const int i0  = blockIdx.y * 16;
  const int z   = blockIdx.z;              // b*NH + h
  const int bb  = z / NH, h = z % NH;

  if (w == 0) {
#pragma unroll
    for (int c = 0; c < BSZ; ++c)
      tdm_load_1d_to_lds(kh16 + (size_t)(c * NH + h) * SEQ * DKH +
                             (size_t)jblk * DKH,
                         lds_addr_of(ldsK[c]), 64 * DKH);
  }
  __builtin_amdgcn_s_wait_tensorcnt(0);
  __syncthreads();

  const _Float16* qslab = qh16 + (size_t)z * SEQ * DKH;

  v8f e[BSZ];
#pragma unroll
  for (int c = 0; c < BSZ; ++c) {
    v8f acc = {};
#pragma unroll
    for (int k0 = 0; k0 < DKH; k0 += 32) {
      v16h a = load_af(qslab + (size_t)i0 * DKH + k0, DKH);
      v16h b = load_af(&ldsK[c][(size_t)j0l * DKH + k0], DKH);  // B^T shape
      acc = wmma_f16(a, b, acc);
    }
#pragma unroll
    for (int r = 0; r < 8; ++r) acc[r] = __expf(acc[r]);   // no max-sub (as ref)
    e[c] = acc;
  }

  v8f diag = e[0];
  if (bb == 1) diag = e[1];
  else if (bb == 2) diag = e[2];
  else if (bb == 3) diag = e[3];

  v8f r;
#pragma unroll
  for (int i = 0; i < 8; ++i) {
    const float denom = 8.0f * (e[0][i] + e[1][i] + e[2][i] + e[3][i]); // sqrt(64)=8
    r[i] = diag[i] / denom;
  }
  store_d_f16(R16 + (size_t)z * SEQ * SEQ + (size_t)i0 * SEQ + jblk + j0l, SEQ, r);
}

// ---------------------------------------------------------------------------
// Kernel 3: score GEMM.  For each (b,c,h): R16[b,h] (512x512) @ vhp[c,h]
// (512x64 packed, TDM-staged whole slab = 64KB) -> f32 score[b,c,h][S][DK].
// One wave per 16-row strip (16x64), 8 waves/block. Grid: (S/128, B*B*H).
// ---------------------------------------------------------------------------
__global__ __launch_bounds__(256) void k_score_gemm(
    const _Float16* __restrict__ R16, const _Float16* __restrict__ vhp,
    float* __restrict__ score) {
  constexpr int KT = SEQ >> 5;             // 16 k-tiles
  __shared__ __align__(32) _Float16 ldsV[KT * 4 * 512];   // 64KB

  const int w   = threadIdx.x >> 5;
  const int i0  = blockIdx.x * 128 + w * 16;
  const int z   = blockIdx.y;              // ((b*4)+c)*8 + h
  const int h   = z & 7;
  const int c   = (z >> 3) & 3;
  const int bb  = z >> 5;

  if (w == 0)
    tdm_load_1d_to_lds(vhp + (size_t)(c * NH + h) * SEQ * DKH,
                       lds_addr_of(ldsV), KT * 4 * 512);
  __builtin_amdgcn_s_wait_tensorcnt(0);
  __syncthreads();

  const _Float16* rslab = R16 + (size_t)(bb * NH + h) * SEQ * SEQ;
  float* oslab = score + (size_t)z * SEQ * DKH;

  v8f acc[4] = {{}, {}, {}, {}};
  for (int kt = 0; kt < KT; ++kt) {
    const int k0 = kt << 5;
    if (kt + 1 < KT)
      __builtin_prefetch(rslab + (size_t)i0 * SEQ + k0 + 32, 0, 1);
    v16h a = load_af(rslab + (size_t)i0 * SEQ + k0, SEQ);
#pragma unroll
    for (int nt = 0; nt < 4; ++nt)
      acc[nt] = wmma_f16(a, load_bp(ldsV, nt * KT + kt), acc[nt]);
  }
#pragma unroll
  for (int nt = 0; nt < 4; ++nt)
    store_d_f32(oslab + (size_t)i0 * DKH + nt * 16, DKH, acc[nt]);
}

// ---------------------------------------------------------------------------
// Kernel 4: per-row (DK=64) softmax + unbiased-std LayerNorm, summed over the
// c-batch, plus BS*qh; writes f16 concat [B,S,DM].  One wave32 per row,
// 2 elements per lane, __shfl_xor reductions.
// ---------------------------------------------------------------------------
static __device__ inline float wred_sum(float v) {
#pragma unroll
  for (int m = 16; m >= 1; m >>= 1) v += __shfl_xor(v, m, 32);
  return v;
}
static __device__ inline float wred_max(float v) {
#pragma unroll
  for (int m = 16; m >= 1; m >>= 1) v = fmaxf(v, __shfl_xor(v, m, 32));
  return v;
}

__global__ __launch_bounds__(256) void k_softmax_ln(
    const float* __restrict__ score, const float* __restrict__ qh32,
    const float* __restrict__ alpha, const float* __restrict__ beta,
    _Float16* __restrict__ concat16) {
  const int row = blockIdx.x * 8 + (threadIdx.x >> 5);   // over B*H*S = 16384
  const int lane = threadIdx.x & 31;
  const int bb = row >> 12;
  const int h  = (row >> 9) & 7;
  const int s  = row & 511;
  const int d0 = lane, d1 = lane + 32;

  const float a0 = alpha[d0], a1 = alpha[d1];
  const float g0 = beta[d0],  g1 = beta[d1];

  float acc0 = 0.f, acc1 = 0.f;
#pragma unroll
  for (int c = 0; c < BSZ; ++c) {
    const size_t base = ((size_t)((bb * 4 + c) * NH + h) * SEQ + s) * DKH;
    const float x0 = score[base + d0];
    const float x1 = score[base + d1];
    const float mx = wred_max(fmaxf(x0, x1));
    const float e0 = __expf(x0 - mx), e1 = __expf(x1 - mx);
    const float inv = 1.0f / wred_sum(e0 + e1);
    const float p0 = e0 * inv, p1 = e1 * inv;
    const float mu = wred_sum(p0 + p1) * (1.0f / 64.0f);
    const float ss = wred_sum(p0 * p0 + p1 * p1);
    const float var = fmaxf((ss - 64.0f * mu * mu) * (1.0f / 63.0f), 0.0f);
    const float inv_std = 1.0f / (sqrtf(var) + EPS);
    acc0 += a0 * (p0 - mu) * inv_std + g0;
    acc1 += a1 * (p1 - mu) * inv_std + g1;
  }
  const size_t qoff = ((size_t)(bb * NH + h) * SEQ + s) * DKH;
  acc0 += (float)BSZ * qh32[qoff + d0];
  acc1 += (float)BSZ * qh32[qoff + d1];

  const size_t coff = ((size_t)(bb * SEQ + s)) * DMODEL + h * DKH;
  concat16[coff + d0] = (_Float16)acc0;
  concat16[coff + d1] = (_Float16)acc1;
}

// ---------------------------------------------------------------------------
// Kernel 5: output GEMM  out = concat16(2048x512) @ Wo(packed, TDM-staged)
// + bo -> f32 row-major.
// ---------------------------------------------------------------------------
__global__ __launch_bounds__(256) void k_gemm_out(
    const _Float16* __restrict__ X, const _Float16* __restrict__ Wp,
    const float* __restrict__ bias, float* __restrict__ out) {
  constexpr int KT = DMODEL >> 5;
  __shared__ __align__(32) _Float16 ldsB[4 * KT * 512];   // 64KB

  const int w  = threadIdx.x >> 5;
  const int m0 = blockIdx.x * 128 + (w & 3) * 32;
  const int n0 = blockIdx.y * 64 + (w >> 2) * 32;
  const int ltile0 = ((w >> 2) * 2) * KT;

  if (w == 0)
    tdm_load_1d_to_lds(Wp + ((size_t)(blockIdx.y * 4) * KT << 9),
                       lds_addr_of(ldsB), 4 * KT * 512);
  __builtin_amdgcn_s_wait_tensorcnt(0);
  __syncthreads();

  v8f acc[2][2] = {{{}, {}}, {{}, {}}};
  for (int kt = 0; kt < KT; ++kt) {
    const int k0 = kt << 5;
    if (kt + 1 < KT) {
      __builtin_prefetch(X + (size_t)m0 * DMODEL + k0 + 32, 0, 1);
      __builtin_prefetch(X + (size_t)(m0 + 16) * DMODEL + k0 + 32, 0, 1);
    }
    v16h a0 = load_af(X + (size_t)m0 * DMODEL + k0, DMODEL);
    v16h a1 = load_af(X + (size_t)(m0 + 16) * DMODEL + k0, DMODEL);
    v16h b0 = load_bp(ldsB, ltile0 + kt);
    v16h b1 = load_bp(ldsB, ltile0 + KT + kt);
    acc[0][0] = wmma_f16(a0, b0, acc[0][0]);
    acc[0][1] = wmma_f16(a0, b1, acc[0][1]);
    acc[1][0] = wmma_f16(a1, b0, acc[1][0]);
    acc[1][1] = wmma_f16(a1, b1, acc[1][1]);
  }

  const int lane = threadIdx.x & 31;
  const int nl   = lane & 15;
  const int mhi  = (lane >> 4) << 3;
#pragma unroll
  for (int i = 0; i < 2; ++i)
#pragma unroll
    for (int j = 0; j < 2; ++j) {
      const int col = n0 + 16 * j + nl;
      const float bv = bias[col];
#pragma unroll
      for (int r = 0; r < 8; ++r) {
        const int row = m0 + 16 * i + mhi + r;
        out[(size_t)row * DMODEL + col] = acc[i][j][r] + bv;
      }
    }
}

// ---------------------------------------------------------------------------
// Host launcher
// ---------------------------------------------------------------------------
extern "C" void kernel_launch(void* const* d_in, const int* in_sizes, int n_in,
                              void* d_out, int out_size, void* d_ws, size_t ws_size,
                              hipStream_t stream) {
  const float* q  = (const float*)d_in[0];
  const float* k  = (const float*)d_in[1];
  const float* v  = (const float*)d_in[2];
  const float* Wq = (const float*)d_in[3];
  const float* bq = (const float*)d_in[4];
  const float* Wk = (const float*)d_in[5];
  const float* bk = (const float*)d_in[6];
  const float* Wv = (const float*)d_in[7];
  const float* bv = (const float*)d_in[8];
  const float* Wo = (const float*)d_in[9];
  const float* bo = (const float*)d_in[10];
  const float* alpha = (const float*)d_in[11];
  const float* beta  = (const float*)d_in[12];
  float* out = (float*)d_out;

  // ---- workspace layout (256B aligned) --------------------------------
  char* p = (char*)d_ws;
  auto take = [&](size_t bytes) {
    char* r = p;
    p += (bytes + 255) & ~(size_t)255;
    return (void*)r;
  };
  const size_t NX = (size_t)MROWS * DMODEL;        // 2048*512
  const size_t NW = (size_t)DMODEL * DMODEL;       // 512*512
  const size_t NHD = (size_t)BSZ * NH * SEQ * DKH; // 1M head elems
  _Float16* q16  = (_Float16*)take(NX * 2);
  _Float16* k16  = (_Float16*)take(NX * 2);
  _Float16* v16  = (_Float16*)take(NX * 2);
  _Float16* wqp  = (_Float16*)take(NW * 2);        // packed B fragments
  _Float16* wkp  = (_Float16*)take(NW * 2);
  _Float16* wvp  = (_Float16*)take(NW * 2);
  _Float16* wop  = (_Float16*)take(NW * 2);
  _Float16* qh16 = (_Float16*)take(NHD * 2);
  _Float16* kh16 = (_Float16*)take(NHD * 2);
  _Float16* vh16 = (_Float16*)take(NHD * 2);
  _Float16* vhp  = (_Float16*)take(NHD * 2);       // packed per-(c,h) slabs
  float*    qh32 = (float*)take(NHD * 4);
  _Float16* R16  = (_Float16*)take((size_t)BSZ * NH * SEQ * SEQ * 2);
  float*    sc32 = (float*)take((size_t)BSZ * BSZ * NH * SEQ * DKH * 4);
  _Float16* c16  = (_Float16*)take(NX * 2);
  (void)ws_size; (void)n_in; (void)in_sizes; (void)out_size;

  // ---- stage 0: activation converts + weight convert/pack -------------
  const int CT = 256;
  k_f32_to_f16<<<(int)((NX + CT - 1) / CT), CT, 0, stream>>>(q, q16, (int)NX);
  k_f32_to_f16<<<(int)((NX + CT - 1) / CT), CT, 0, stream>>>(k, k16, (int)NX);
  k_f32_to_f16<<<(int)((NX + CT - 1) / CT), CT, 0, stream>>>(v, v16, (int)NX);
  const int WB = (int)((NW + CT - 1) / CT);
  k_pack_b_f32<<<WB, CT, 0, stream>>>(Wq, wqp, DMODEL, DMODEL);
  k_pack_b_f32<<<WB, CT, 0, stream>>>(Wk, wkp, DMODEL, DMODEL);
  k_pack_b_f32<<<WB, CT, 0, stream>>>(Wv, wvp, DMODEL, DMODEL);
  k_pack_b_f32<<<WB, CT, 0, stream>>>(Wo, wop, DMODEL, DMODEL);

  // ---- stage 1: projections (WMMA GEMM + head-split epilogue) ---------
  dim3 gGemm(MROWS / 128, DMODEL / 64);
  k_gemm_proj_head<true ><<<gGemm, 256, 0, stream>>>(q16, wqp, bq, qh16, qh32);
  k_gemm_proj_head<false><<<gGemm, 256, 0, stream>>>(k16, wkp, bk, kh16, nullptr);
  k_gemm_proj_head<false><<<gGemm, 256, 0, stream>>>(v16, wvp, bv, vh16, nullptr);

  // ---- stage 1b: repack vh slabs into B-fragment layout ---------------
  dim3 gPackV((SEQ * DKH) / CT, BSZ * NH);
  k_pack_b_f16<<<gPackV, CT, 0, stream>>>(vh16, vhp, SEQ, DKH, (size_t)SEQ * DKH);

  // ---- stage 2: pairwise exp-score ratio R ----------------------------
  dim3 gR(SEQ / 64, SEQ / 16, BSZ * NH);
  k_pairwise_R<<<gR, 128, 0, stream>>>(qh16, kh16, R16);

  // ---- stage 3: score = R @ V per (b,c,h) -----------------------------
  dim3 gS(SEQ / 128, BSZ * BSZ * NH);
  k_score_gemm<<<gS, 256, 0, stream>>>(R16, vhp, sc32);

  // ---- stage 4: softmax + LN + sum_c + BS*q ---------------------------
  k_softmax_ln<<<(BSZ * NH * SEQ) / 8, 256, 0, stream>>>(sc32, qh32, alpha, beta, c16);

  // ---- stage 5: output projection -------------------------------------
  k_gemm_out<<<gGemm, 256, 0, stream>>>(c16, wop, bo, out);
}
